// PixelAttention_73830487818432
// MI455X (gfx1250) — matrix operand order repaired and checked by reading, not compiled, not run on hardware
//
#include <hip/hip_runtime.h>
#include <hip/hip_bf16.h>

// ---------------------------------------------------------------------------
// PixelAttention on MI455X (gfx1250, wave32):
//   - QKV projection + 3x3 convs as implicit GEMMs via v_wmma_f32_16x16x32_bf16
//   - register-blocked tiles (2x2 for convs, 1x12 for QKV) to cut vmem/WMMA
//   - attention scores/softmax on VALU (one wave per pixel)
// ---------------------------------------------------------------------------

typedef __attribute__((ext_vector_type(16))) __bf16 v16bf;
typedef __attribute__((ext_vector_type(8)))  float  v8f;
typedef __attribute__((ext_vector_type(4)))  float  f4v;
typedef unsigned short ushort_t;

union ABFrag { v16bf v; f4v f[2]; };

__device__ __forceinline__ ushort_t f2bf(float f) {
    unsigned int u = __float_as_uint(f);
    unsigned int r = u + 0x7FFFu + ((u >> 16) & 1u);   // round-to-nearest-even
    return (ushort_t)(r >> 16);
}
__device__ __forceinline__ float bf2f(ushort_t u) {
    return __uint_as_float(((unsigned int)u) << 16);
}

// Problem constants
static constexpr int BB = 8, HH = 128, WW = 128, DD = 64;
static constexpr int NPIX = BB * HH * WW;          // 131072
static constexpr int HWSZ = HH * WW;               // 16384

// ---------------------------------------------------------------------------
// 1) x (f32) -> bf16
// ---------------------------------------------------------------------------
__global__ __launch_bounds__(256) void cvt_bf16_kernel(const float* __restrict__ src,
                                                       ushort_t* __restrict__ dst,
                                                       int total4) {
    int idx = blockIdx.x * blockDim.x + threadIdx.x;
    if (idx >= total4) return;
    f4v v = reinterpret_cast<const f4v*>(src)[idx];
    int o = idx * 4;
    dst[o + 0] = f2bf(v.x);
    dst[o + 1] = f2bf(v.y);
    dst[o + 2] = f2bf(v.z);
    dst[o + 3] = f2bf(v.w);
}

// ---------------------------------------------------------------------------
// 2) Pack a [KC*32, Cout] f32 weight matrix into the WMMA B-fragment register
//    image: frag(kc,nt) -> 32 lanes x 16 bf16 contiguous per lane.
//    B layout (16-bit, 32x16): lanes 0-15 = cols 0-15 holding K 0..7,16..23,
//    lanes 16-31 = cols 0-15 holding K 8..15,24..31 (per ISA 7.12.4 pattern).
// ---------------------------------------------------------------------------
__global__ __launch_bounds__(256) void pack_b_kernel(const float* __restrict__ W,
                                                     ushort_t* __restrict__ out,
                                                     int KC, int NT, int Cout) {
    int idx = blockIdx.x * blockDim.x + threadIdx.x;
    int total = KC * NT * 512;
    if (idx >= total) return;
    int i    = idx & 15;
    int lane = (idx >> 4) & 31;
    int frag = idx >> 9;
    int kc = frag / NT;
    int nt = frag - kc * NT;
    int k  = kc * 32 + ((i >> 3) * 16 + (lane >> 4) * 8 + (i & 7));
    int n  = nt * 16 + (lane & 15);
    out[idx] = f2bf(W[(size_t)k * Cout + n]);
}

// ---------------------------------------------------------------------------
// 3) QKV projection: [NPIX,64] x (3 x [64,64]) -> q,k,v in bf16.
//    One wave per 16-pixel tile; A fragment reused across all 12 column
//    tiles (Q:0-3, K:4-7, V:8-11)  ->  ~2 b128 loads per WMMA.
// ---------------------------------------------------------------------------
__global__ __launch_bounds__(256) void qkv_gemm_kernel(const ushort_t* __restrict__ xb,
                                                       const ushort_t* __restrict__ wqkv,
                                                       ushort_t* __restrict__ qo,
                                                       ushort_t* __restrict__ ko,
                                                       ushort_t* __restrict__ vo) {
    int tid   = threadIdx.x;
    int lane  = tid & 31;
    int mtile = blockIdx.x * (blockDim.x >> 5) + (tid >> 5);

    int m    = lane & 15;
    int half = lane >> 4;
    int kb   = half * 8;
    int p    = mtile * 16 + m;

    v8f acc[12];
#pragma unroll
    for (int j = 0; j < 12; ++j) acc[j] = (v8f){};

    const ushort_t* arow = xb + (size_t)p * DD + kb;
#pragma unroll
    for (int c = 0; c < 2; ++c) {
        ABFrag a;
        a.f[0] = *reinterpret_cast<const f4v*>(arow + c * 32);
        a.f[1] = *reinterpret_cast<const f4v*>(arow + c * 32 + 16);
#pragma unroll
        for (int j = 0; j < 12; ++j) {
            int mat = j >> 2;
            int nt  = j & 3;
            ABFrag b;
            const ushort_t* bp =
                wqkv + (size_t)mat * 4096 + ((size_t)((c * 4 + nt) * 32 + lane)) * 16;
            b.f[0] = *reinterpret_cast<const f4v*>(bp);
            b.f[1] = *reinterpret_cast<const f4v*>(bp + 8);
            acc[j] = __builtin_amdgcn_wmma_f32_16x16x32_bf16(
                        false, a.v, false, b.v, (short)0, acc[j], false, false);
        }
    }

#pragma unroll
    for (int j = 0; j < 12; ++j) {
        int mat = j >> 2;
        int nt  = j & 3;
        ushort_t* dst = (mat == 0) ? qo : (mat == 1) ? ko : vo;
        int ch = nt * 16 + (lane & 15);
#pragma unroll
        for (int r = 0; r < 8; ++r) {
            int pr = mtile * 16 + r + half * 8;
            dst[(size_t)pr * DD + ch] = f2bf(acc[j][r]);
        }
    }
}

// ---------------------------------------------------------------------------
// 4) Attention: one wave per pixel, lane owns 2 channels.
//    Faithful semantics: score = (q.k)*mask (OOB -> 0), softmax over all 9,
//    v of OOB neighbors is 0.  t = x + attn; store f32 + bf16.
// ---------------------------------------------------------------------------
__global__ __launch_bounds__(256) void attn_kernel(const float* __restrict__ x,
                                                   const ushort_t* __restrict__ qb,
                                                   const ushort_t* __restrict__ kb,
                                                   const ushort_t* __restrict__ vb,
                                                   float* __restrict__ tf,
                                                   ushort_t* __restrict__ tb) {
    int tid  = threadIdx.x;
    int lane = tid & 31;
    int p    = blockIdx.x * (blockDim.x >> 5) + (tid >> 5);
    int b  = p >> 14;
    int hw = p & (HWSZ - 1);
    int h  = hw >> 7;
    int w  = hw & (WW - 1);
    int c  = lane * 2;

    float q0 = bf2f(qb[(size_t)p * DD + c]);
    float q1 = bf2f(qb[(size_t)p * DD + c + 1]);

    float sc[9], v0[9], v1[9];
    int j = 0;
#pragma unroll
    for (int di = -1; di <= 1; ++di) {
#pragma unroll
        for (int dj = -1; dj <= 1; ++dj) {
            int hh = h + di, ww = w + dj;
            bool valid = (hh >= 0) && (hh < HH) && (ww >= 0) && (ww < WW);
            float s = 0.0f, vv0 = 0.0f, vv1 = 0.0f;
            if (valid) {   // wave-uniform branch (one pixel per wave)
                int np = (b << 14) + (hh << 7) + ww;
                float k0 = bf2f(kb[(size_t)np * DD + c]);
                float k1 = bf2f(kb[(size_t)np * DD + c + 1]);
                float part = q0 * k0 + q1 * k1;
#pragma unroll
                for (int off = 16; off > 0; off >>= 1)
                    part += __shfl_xor(part, off, 32);
                s = part;
                vv0 = bf2f(vb[(size_t)np * DD + c]);
                vv1 = bf2f(vb[(size_t)np * DD + c + 1]);
            }
            sc[j] = s; v0[j] = vv0; v1[j] = vv1; ++j;
        }
    }

    float mx = sc[0];
#pragma unroll
    for (int t = 1; t < 9; ++t) mx = fmaxf(mx, sc[t]);
    float den = 0.0f, al[9];
#pragma unroll
    for (int t = 0; t < 9; ++t) { al[t] = __expf(sc[t] - mx); den += al[t]; }
    float inv = 1.0f / den;
    float a0 = 0.0f, a1 = 0.0f;
#pragma unroll
    for (int t = 0; t < 9; ++t) { a0 += al[t] * v0[t]; a1 += al[t] * v1[t]; }
    a0 *= inv; a1 *= inv;

    float t0 = x[(size_t)p * DD + c]     + a0;
    float t1 = x[(size_t)p * DD + c + 1] + a1;
    tf[(size_t)p * DD + c]     = t0;
    tf[(size_t)p * DD + c + 1] = t1;
    tb[(size_t)p * DD + c]     = f2bf(t0);
    tb[(size_t)p * DD + c + 1] = f2bf(t1);
}

// ---------------------------------------------------------------------------
// 5) 3x3 conv as implicit GEMM with 2x2 register blocking:
//    one wave per 32(pixels) x 32(outch) macro-tile.  Per (tap, kchunk):
//    2 A frags + 2 B frags -> 4 WMMAs (2 b128 loads per WMMA).
//    Bias + ReLU fused.  FUSE_RES: out_f32 = resid + relu(conv).
// ---------------------------------------------------------------------------
template <int CIN, int COUT, bool FUSE_RES>
__global__ __launch_bounds__(256) void conv3x3_wmma_kernel(
        const ushort_t* __restrict__ src,
        const ushort_t* __restrict__ wpack,
        const float* __restrict__ bias,
        const float* __restrict__ resid,
        ushort_t* __restrict__ dst_bf,
        float* __restrict__ dst_f32) {
    constexpr int KC  = CIN / 32;
    constexpr int NT  = COUT / 16;
    constexpr int NT2 = COUT / 32;

    int tid   = threadIdx.x;
    int lane  = tid & 31;
    int gwave = blockIdx.x * (blockDim.x >> 5) + (tid >> 5);
    int mt2   = gwave / NT2;
    int nt2   = gwave - mt2 * NT2;

    int m    = lane & 15;
    int half = lane >> 4;
    int kb   = half * 8;
    // 32 consecutive pixels: 32 | WW, both 16-row tiles sit in one image row
    int p0 = mt2 * 32 + m;
    int p1 = p0 + 16;
    int b  = p0 >> 14;
    int hw = p0 & (HWSZ - 1);
    int h  = hw >> 7;
    int w0 = hw & (WW - 1);
    int w1 = w0 + 16;

    v8f acc[2][2];
#pragma unroll
    for (int ti = 0; ti < 2; ++ti)
#pragma unroll
        for (int tj = 0; tj < 2; ++tj) acc[ti][tj] = (v8f){};

#pragma unroll
    for (int di = -1; di <= 1; ++di) {
#pragma unroll
        for (int dj = -1; dj <= 1; ++dj) {
            int t  = (di + 1) * 3 + (dj + 1);
            int hh = h + di;
            int wa = w0 + dj, wb_ = w1 + dj;
            bool hok = (hh >= 0) && (hh < HH);
            bool va  = hok && (wa >= 0) && (wa < WW);
            bool vb_ = hok && (wb_ >= 0) && (wb_ < WW);
#pragma unroll
            for (int c = 0; c < KC; ++c) {
                ABFrag a0, a1, bf0, bf1;
                f4v z = {0.0f, 0.0f, 0.0f, 0.0f};
                if (va) {
                    int np = (b << 14) + (hh << 7) + wa;
                    const ushort_t* ar = src + (size_t)np * CIN + kb + c * 32;
                    a0.f[0] = *reinterpret_cast<const f4v*>(ar);
                    a0.f[1] = *reinterpret_cast<const f4v*>(ar + 16);
                } else { a0.f[0] = z; a0.f[1] = z; }
                if (vb_) {
                    int np = (b << 14) + (hh << 7) + wb_;
                    const ushort_t* ar = src + (size_t)np * CIN + kb + c * 32;
                    a1.f[0] = *reinterpret_cast<const f4v*>(ar);
                    a1.f[1] = *reinterpret_cast<const f4v*>(ar + 16);
                } else { a1.f[0] = z; a1.f[1] = z; }

                const ushort_t* bp0 =
                    wpack + ((size_t)((t * KC + c) * NT + nt2 * 2) * 32 + lane) * 16;
                const ushort_t* bp1 = bp0 + 512;   // next ntile fragment
                bf0.f[0] = *reinterpret_cast<const f4v*>(bp0);
                bf0.f[1] = *reinterpret_cast<const f4v*>(bp0 + 8);
                bf1.f[0] = *reinterpret_cast<const f4v*>(bp1);
                bf1.f[1] = *reinterpret_cast<const f4v*>(bp1 + 8);

                acc[0][0] = __builtin_amdgcn_wmma_f32_16x16x32_bf16(
                        false, a0.v, false, bf0.v, (short)0, acc[0][0], false, false);
                acc[0][1] = __builtin_amdgcn_wmma_f32_16x16x32_bf16(
                        false, a0.v, false, bf1.v, (short)0, acc[0][1], false, false);
                acc[1][0] = __builtin_amdgcn_wmma_f32_16x16x32_bf16(
                        false, a1.v, false, bf0.v, (short)0, acc[1][0], false, false);
                acc[1][1] = __builtin_amdgcn_wmma_f32_16x16x32_bf16(
                        false, a1.v, false, bf1.v, (short)0, acc[1][1], false, false);
            }
        }
    }

#pragma unroll
    for (int tj = 0; tj < 2; ++tj) {
        int ch = (nt2 * 2 + tj) * 16 + (lane & 15);
        float bv = bias[ch];
#pragma unroll
        for (int ti = 0; ti < 2; ++ti) {
#pragma unroll
            for (int r = 0; r < 8; ++r) {
                int pr = mt2 * 32 + ti * 16 + r + half * 8;
                float val = fmaxf(acc[ti][tj][r] + bv, 0.0f);
                if (FUSE_RES) {
                    dst_f32[(size_t)pr * COUT + ch] =
                        resid[(size_t)pr * COUT + ch] + val;
                } else {
                    dst_bf[(size_t)pr * COUT + ch] = f2bf(val);
                }
            }
        }
    }
}

// ---------------------------------------------------------------------------
// Launch
// ---------------------------------------------------------------------------
extern "C" void kernel_launch(void* const* d_in, const int* in_sizes, int n_in,
                              void* d_out, int out_size, void* d_ws, size_t ws_size,
                              hipStream_t stream) {
    const float* x   = (const float*)d_in[0];
    const float* Q_P = (const float*)d_in[1];
    const float* K_P = (const float*)d_in[2];
    const float* V_P = (const float*)d_in[3];
    const float* W1  = (const float*)d_in[4];
    const float* B1  = (const float*)d_in[5];
    const float* W2  = (const float*)d_in[6];
    const float* B2  = (const float*)d_in[7];
    const float* W3  = (const float*)d_in[8];
    const float* B3  = (const float*)d_in[9];
    float* out = (float*)d_out;

    char* ws = (char*)d_ws;
    // Workspace layout (bytes), with aliasing of dead buffers:
    constexpr size_t SZ_BF  = (size_t)NPIX * DD * 2;      // 16 MB
    constexpr size_t SZ_F32 = (size_t)NPIX * DD * 4;      // 32 MB
    const size_t OFF_X16  = 0;
    const size_t OFF_QB   = OFF_X16 + SZ_BF;
    const size_t OFF_KB   = OFF_QB + SZ_BF;
    const size_t OFF_VB   = OFF_KB + SZ_BF;
    const size_t OFF_TF   = OFF_VB + SZ_BF;
    const size_t OFF_TB   = OFF_TF + SZ_F32;
    const size_t OFF_U1   = OFF_QB;            // alias (q dead after attention)
    const size_t OFF_U2   = OFF_KB;            // alias (k,v dead), needs 2*SZ_BF
    const size_t OFF_WQKV = OFF_TB + SZ_BF;
    const size_t OFF_W1P  = OFF_WQKV + 3 * 4096 * 2;
    const size_t OFF_W2P  = OFF_W1P + 36864 * 2;
    const size_t OFF_W3P  = OFF_W2P + 73728 * 2;

    ushort_t* x16  = (ushort_t*)(ws + OFF_X16);
    ushort_t* qb   = (ushort_t*)(ws + OFF_QB);
    ushort_t* kb   = (ushort_t*)(ws + OFF_KB);
    ushort_t* vb   = (ushort_t*)(ws + OFF_VB);
    float*    tf   = (float*)   (ws + OFF_TF);
    ushort_t* tb   = (ushort_t*)(ws + OFF_TB);
    ushort_t* u1   = (ushort_t*)(ws + OFF_U1);
    ushort_t* u2   = (ushort_t*)(ws + OFF_U2);
    ushort_t* wqkv = (ushort_t*)(ws + OFF_WQKV);
    ushort_t* w1p  = (ushort_t*)(ws + OFF_W1P);
    ushort_t* w2p  = (ushort_t*)(ws + OFF_W2P);
    ushort_t* w3p  = (ushort_t*)(ws + OFF_W3P);

    const int threads = 256;

    // 1) x -> bf16
    {
        int total4 = NPIX * DD / 4;
        cvt_bf16_kernel<<<(total4 + threads - 1) / threads, threads, 0, stream>>>(
            x, x16, total4);
    }
    // 2) pack weights into B-fragment layout
    pack_b_kernel<<<16,  threads, 0, stream>>>(Q_P, wqkv,          2, 4, 64);
    pack_b_kernel<<<16,  threads, 0, stream>>>(K_P, wqkv + 4096,   2, 4, 64);
    pack_b_kernel<<<16,  threads, 0, stream>>>(V_P, wqkv + 8192,   2, 4, 64);
    pack_b_kernel<<<144, threads, 0, stream>>>(W1,  w1p,          18, 4, 64);
    pack_b_kernel<<<288, threads, 0, stream>>>(W2,  w2p,          18, 8, 128);
    pack_b_kernel<<<288, threads, 0, stream>>>(W3,  w3p,          36, 4, 64);

    // 3) QKV projection (one wave per 16-pixel tile, 12 column tiles)
    qkv_gemm_kernel<<<(NPIX / 16) / 8, threads, 0, stream>>>(x16, wqkv, qb, kb, vb);

    // 4) attention (one wave per pixel)
    attn_kernel<<<NPIX / 8, threads, 0, stream>>>(x, qb, kb, vb, tf, tb);

    // 5) conv stack (2x2 blocked macro-tiles: 32 pixels x 32 channels / wave)
    conv3x3_wmma_kernel<64, 64, false><<<(NPIX / 32) * 2 / 8, threads, 0, stream>>>(
        tb, w1p, B1, nullptr, u1, nullptr);
    conv3x3_wmma_kernel<64, 128, false><<<(NPIX / 32) * 4 / 8, threads, 0, stream>>>(
        u1, w2p, B2, nullptr, u2, nullptr);
    conv3x3_wmma_kernel<128, 64, true><<<(NPIX / 32) * 2 / 8, threads, 0, stream>>>(
        u2, w3p, B3, tf, nullptr, out);
}
